// TransformerEncoder_901943132508
// MI455X (gfx1250) — compile-verified, matrix operand order checked by reading, not run on hardware
//
#include <hip/hip_runtime.h>
#include <hip/hip_bf16.h>

typedef __attribute__((ext_vector_type(16))) _Float16 v16h;
typedef __attribute__((ext_vector_type(8)))  _Float16 v8h;
typedef __attribute__((ext_vector_type(8)))  float    v8f;

// ---------------------------------------------------------------------------
// Fragment loader for V_WMMA_F32_16X16X32_F16 operands.
// LDS tile is row-major [row][k] with row stride LDSTR halves (144B, 16B-mult).
// Per ISA 7.12.2 (16-bit A-matrix 16x32): lanes 0-15 hold K=0..7 then 16..23,
// lanes 16-31 hold K=8..15 then 24..31.  B operand uses the mirrored layout
// with lane->N, so storing B transposed ([n][k]) makes both loads identical:
// two contiguous 16-byte LDS reads per lane.
// ---------------------------------------------------------------------------
__device__ __forceinline__ v16h load_frag(const _Float16* p /*row base + kb*/, int hi) {
  const v8h* pv = (const v8h*)p;
  v8h lo = pv[hi];      // K = hi*8 .. hi*8+7
  v8h hh = pv[hi + 2];  // K = hi*8+16 .. hi*8+23
  return __builtin_shufflevector(lo, hh, 0,1,2,3,4,5,6,7,8,9,10,11,12,13,14,15);
}

__device__ __forceinline__ v8f wmma_f16(v16h a, v16h b, v8f c) {
  return __builtin_amdgcn_wmma_f32_16x16x32_f16(false, a, false, b, (short)0, c, false, false);
}

#define LDSTR 72   // 64 + 8 halves pad: kills bank conflicts, keeps 16B align

// ---------------------------------------------------------------------------
// f32 -> f16 weight conversion
// ---------------------------------------------------------------------------
__global__ void cvt_f32_f16(const float* __restrict__ in, _Float16* __restrict__ out, size_t n) {
  size_t i = (size_t)blockIdx.x * blockDim.x + threadIdx.x;
  size_t stride = (size_t)gridDim.x * blockDim.x;
  for (; i < n; i += stride) out[i] = (_Float16)in[i];
}

// ---------------------------------------------------------------------------
// Embedding lookup + sinusoidal positional encoding -> x (f32 + f16)
// ---------------------------------------------------------------------------
__global__ __launch_bounds__(256) void embed_pe(const int* __restrict__ tokens,
                                                const float* __restrict__ emb,
                                                float* __restrict__ xf,
                                                _Float16* __restrict__ xh,
                                                int S, int D) {
  int row = blockIdx.x;
  int s = row % S;
  int tok = tokens[row];
  float c = -logf(10000.f) / (float)D;
  for (int d = threadIdx.x; d < D; d += blockDim.x) {
    float freq = __expf((float)(2 * (d >> 1)) * c);
    float ang = (float)s * freq;
    float pe = (d & 1) ? cosf(ang) : sinf(ang);
    float v = emb[(size_t)tok * D + d] + pe;
    xf[(size_t)row * D + d] = v;
    xh[(size_t)row * D + d] = (_Float16)v;
  }
}

// ---------------------------------------------------------------------------
// Tiled WMMA GEMM: C[M,N] = A[M,K](f16,row) * W[K,N](f16,row) + bias (+ReLU)
// 128x128 block tile, K-tile 64.  8 waves in a 2(M) x 4(N) grid; each wave
// owns a 64x32 sub-tile = 4x2 accumulators.  Per k-step: 6 fragment loads
// feed 8 WMMAs.  Compile-time RELU / f32-out / f16-out selection keeps the
// epilogue branch-free.
// ---------------------------------------------------------------------------
template<int RELU, int WF, int WH>
__global__ __launch_bounds__(256) void gemm_wmma(const _Float16* __restrict__ A,
                                                 const _Float16* __restrict__ W,
                                                 const float* __restrict__ bias,
                                                 float* __restrict__ Cf,
                                                 _Float16* __restrict__ Ch,
                                                 int M, int N, int K) {
  __shared__ alignas(16) _Float16 As[128][LDSTR];   // [m][k]
  __shared__ alignas(16) _Float16 Bs[128][LDSTR];   // transposed: [n][k]
  const int t = threadIdx.x;
  const int lane = t & 31, w = t >> 5;
  const int hi = lane >> 4, lc = lane & 15;
  const int m0 = blockIdx.y * 128;
  const int n0 = blockIdx.x * 128;
  const int wm = (w & 1) * 64;    // wave M base within block tile
  const int wn = (w >> 1) * 32;   // wave N base within block tile

  v8f acc[4][2];
  #pragma unroll
  for (int i = 0; i < 4; ++i)
    #pragma unroll
    for (int j = 0; j < 2; ++j)
      #pragma unroll
      for (int r = 0; r < 8; ++r) acc[i][j][r] = 0.f;

  const int arow = t >> 1, acg = (t & 1) * 32;   // A: 128 rows x 64 k, 32 halves/thr
  const int bn = t & 127, bkg = (t >> 7) * 32;   // B transpose-gather, 32 halves/thr

  for (int k0 = 0; k0 < K; k0 += 64) {
    const _Float16* ag = A + (size_t)(m0 + arow) * K + k0 + acg;
    #pragma unroll
    for (int j = 0; j < 32; j += 8)
      *(v8h*)&As[arow][acg + j] = *(const v8h*)(ag + j);

    const _Float16* bg = W + (size_t)(k0 + bkg) * N + n0 + bn;
    #pragma unroll
    for (int jj = 0; jj < 32; jj += 8) {
      v8h tv;
      #pragma unroll
      for (int i = 0; i < 8; ++i) tv[i] = bg[(size_t)(jj + i) * N];
      *(v8h*)&Bs[bn][bkg + jj] = tv;
    }
    if (k0 + 64 < K) {  // hint next K-tile into cache (global_prefetch_b8)
      __builtin_prefetch(ag + 64, 0, 0);
      __builtin_prefetch(bg + (size_t)64 * N, 0, 0);
    }
    __syncthreads();
    #pragma unroll
    for (int kb = 0; kb < 64; kb += 32) {
      v16h af[4], bf[2];
      #pragma unroll
      for (int i = 0; i < 4; ++i) af[i] = load_frag(&As[wm + i * 16 + lc][kb], hi);
      #pragma unroll
      for (int j = 0; j < 2; ++j) bf[j] = load_frag(&Bs[wn + j * 16 + lc][kb], hi);
      #pragma unroll
      for (int i = 0; i < 4; ++i)
        #pragma unroll
        for (int j = 0; j < 2; ++j)
          acc[i][j] = wmma_f16(af[i], bf[j], acc[i][j]);
    }
    __syncthreads();
  }

  #pragma unroll
  for (int j = 0; j < 2; ++j) {
    const int col = n0 + wn + j * 16 + lc;
    const float bcol = bias[col];
    #pragma unroll
    for (int i = 0; i < 4; ++i) {
      #pragma unroll
      for (int r = 0; r < 8; ++r) {
        int row = m0 + wm + i * 16 + r + hi * 8;
        float v = acc[i][j][r] + bcol;
        if constexpr (RELU) v = fmaxf(v, 0.f);
        size_t idx = (size_t)row * N + col;
        if constexpr (WF) Cf[idx] = v;
        if constexpr (WH) Ch[idx] = (_Float16)v;
      }
    }
  }
}

// ---------------------------------------------------------------------------
// Flash attention: one block = (b, h, 128-query tile); head_dim = 64.
// Scores & P*V via WMMA; online softmax with half-wave shfl reductions.
// C layout: lanes 0-15 own rows 0-7 (vgpr r -> row r), lanes 16-31 rows 8-15.
// ---------------------------------------------------------------------------
__global__ __launch_bounds__(256) void attn_flash(const _Float16* __restrict__ Q,
                                                  const _Float16* __restrict__ Kk,
                                                  const _Float16* __restrict__ V,
                                                  _Float16* __restrict__ O,
                                                  int B, int S, int H, int D) {
  __shared__ alignas(16) _Float16 Qs[128][LDSTR];      // [q][d]
  __shared__ alignas(16) _Float16 Ks[64][LDSTR];       // [key][d] == B^T for QK^T
  __shared__ alignas(16) _Float16 Vt[64][LDSTR];       // transposed: [d][key]
  __shared__ alignas(16) _Float16 Ps[8][16][LDSTR];    // per-wave P tile [q][key]

  const int qtiles = S / 128;
  const int bh = blockIdx.x / qtiles, qt = blockIdx.x % qtiles;
  const int b = bh / H, h = bh % H;
  const int q0 = qt * 128;
  const int t = threadIdx.x, lane = t & 31, w = t >> 5;
  const int hi = lane >> 4, lc = lane & 15;
  const size_t hd = (size_t)h * 64;

  { // load Q tile once: 128 x 64, 32 halves / thread
    int row = t >> 1, cg = (t & 1) * 32;
    const _Float16* g = Q + ((size_t)(b * S + q0 + row)) * D + hd + cg;
    #pragma unroll
    for (int j = 0; j < 32; j += 8)
      *(v8h*)&Qs[row][cg + j] = *(const v8h*)(g + j);
  }

  float mrow[8], lrow[8];
  v8f oacc[4];
  #pragma unroll
  for (int r = 0; r < 8; ++r) { mrow[r] = -1e30f; lrow[r] = 0.f; }
  #pragma unroll
  for (int d = 0; d < 4; ++d)
    #pragma unroll
    for (int r = 0; r < 8; ++r) oacc[d][r] = 0.f;

  for (int k0 = 0; k0 < S; k0 += 64) {
    __syncthreads();  // previous P*V done; Q visible on first iteration
    { // K tile: 64 x 64 row-major (already the transposed-B layout for QK^T)
      int row = t >> 2, cg = (t & 3) * 16;
      const _Float16* g = Kk + ((size_t)(b * S + k0 + row)) * D + hd + cg;
      *(v8h*)&Ks[row][cg]     = *(const v8h*)(g);
      *(v8h*)&Ks[row][cg + 8] = *(const v8h*)(g + 8);
      if (k0 + 64 < S) __builtin_prefetch(g + (size_t)64 * D, 0, 0);
    }
    { // V tile transposed: Vt[d][key]
      int d = t & 63, kg = (t >> 6) * 16;
      v8h t0, t1;
      #pragma unroll
      for (int i = 0; i < 8; ++i) {
        t0[i] = V[((size_t)(b * S + k0 + kg + i))     * D + hd + d];
        t1[i] = V[((size_t)(b * S + k0 + kg + 8 + i)) * D + hd + d];
      }
      *(v8h*)&Vt[d][kg]     = t0;
      *(v8h*)&Vt[d][kg + 8] = t1;
    }
    __syncthreads();

    // scores: 16 q-rows per wave x 64 keys, K-dim = head_dim 64
    v8f sc[4];
    #pragma unroll
    for (int nt = 0; nt < 4; ++nt)
      #pragma unroll
      for (int r = 0; r < 8; ++r) sc[nt][r] = 0.f;
    #pragma unroll
    for (int kb = 0; kb < 64; kb += 32) {
      v16h a = load_frag(&Qs[w * 16 + lc][kb], hi);
      #pragma unroll
      for (int nt = 0; nt < 4; ++nt) {
        v16h bf = load_frag(&Ks[nt * 16 + lc][kb], hi);
        sc[nt] = wmma_f16(a, bf, sc[nt]);
      }
    }
    #pragma unroll
    for (int nt = 0; nt < 4; ++nt)
      #pragma unroll
      for (int r = 0; r < 8; ++r) sc[nt][r] *= 0.125f;  // 1/sqrt(64)

    // online softmax: row stats replicated across each half-wave
    float rmax[8], rsum[8];
    #pragma unroll
    for (int r = 0; r < 8; ++r) {
      float v = sc[0][r];
      #pragma unroll
      for (int nt = 1; nt < 4; ++nt) v = fmaxf(v, sc[nt][r]);
      rmax[r] = v;
    }
    #pragma unroll
    for (int mask = 1; mask < 16; mask <<= 1)
      #pragma unroll
      for (int r = 0; r < 8; ++r)
        rmax[r] = fmaxf(rmax[r], __shfl_xor(rmax[r], mask, 16));
    #pragma unroll
    for (int r = 0; r < 8; ++r) {
      float mn = fmaxf(mrow[r], rmax[r]);
      float al = __expf(mrow[r] - mn);
      mrow[r] = mn;
      lrow[r] *= al;
      rsum[r] = 0.f;
      #pragma unroll
      for (int nt = 0; nt < 4; ++nt) oacc[nt][r] *= al;
    }
    #pragma unroll
    for (int nt = 0; nt < 4; ++nt)
      #pragma unroll
      for (int r = 0; r < 8; ++r) {
        float e = __expf(sc[nt][r] - mrow[r]);
        rsum[r] += e;
        Ps[w][r + hi * 8][nt * 16 + lc] = (_Float16)e;
      }
    #pragma unroll
    for (int mask = 1; mask < 16; mask <<= 1)
      #pragma unroll
      for (int r = 0; r < 8; ++r)
        rsum[r] += __shfl_xor(rsum[r], mask, 16);
    #pragma unroll
    for (int r = 0; r < 8; ++r) lrow[r] += rsum[r];
    __syncthreads();  // P visible; everyone done reading Ks

    // O += P * V  (K-dim = 64 keys)
    #pragma unroll
    for (int kb = 0; kb < 64; kb += 32) {
      v16h a = load_frag(&Ps[w][lc][kb], hi);
      #pragma unroll
      for (int dt = 0; dt < 4; ++dt) {
        v16h bf = load_frag(&Vt[dt * 16 + lc][kb], hi);
        oacc[dt] = wmma_f16(a, bf, oacc[dt]);
      }
    }
  }

  #pragma unroll
  for (int dt = 0; dt < 4; ++dt)
    #pragma unroll
    for (int r = 0; r < 8; ++r) {
      int row = q0 + w * 16 + r + hi * 8;
      int col = (int)hd + dt * 16 + lc;
      O[((size_t)(b * S + row)) * D + col] = (_Float16)(oacc[dt][r] / lrow[r]);
    }
}

// ---------------------------------------------------------------------------
// Fused residual-add + LayerNorm, one block per row (D = 1024, 4 elems/thread)
// ---------------------------------------------------------------------------
__global__ __launch_bounds__(256) void residual_ln(const float* __restrict__ a,
                                                   const float* __restrict__ bres,
                                                   const float* __restrict__ g,
                                                   const float* __restrict__ be,
                                                   float* __restrict__ of,
                                                   _Float16* __restrict__ oh,
                                                   int D) {
  __shared__ float red[256];
  const int row = blockIdx.x, t = threadIdx.x;
  float vals[4];
  float s = 0.f, s2 = 0.f;
  #pragma unroll
  for (int i = 0; i < 4; ++i) {
    int d = t + i * 256;
    float v = a[(size_t)row * D + d] + (bres ? bres[(size_t)row * D + d] : 0.f);
    vals[i] = v; s += v; s2 += v * v;
  }
  red[t] = s; __syncthreads();
  for (int off = 128; off > 0; off >>= 1) { if (t < off) red[t] += red[t + off]; __syncthreads(); }
  float mean = red[0] / (float)D; __syncthreads();
  red[t] = s2; __syncthreads();
  for (int off = 128; off > 0; off >>= 1) { if (t < off) red[t] += red[t + off]; __syncthreads(); }
  float var = red[0] / (float)D - mean * mean;
  float rstd = rsqrtf(var + 1e-5f);
  #pragma unroll
  for (int i = 0; i < 4; ++i) {
    int d = t + i * 256;
    float y = (vals[i] - mean) * rstd * g[d] + be[d];
    if (of) of[(size_t)row * D + d] = y;
    if (oh) oh[(size_t)row * D + d] = (_Float16)y;
  }
}

// ---------------------------------------------------------------------------
// Host orchestration
// ---------------------------------------------------------------------------
extern "C" void kernel_launch(void* const* d_in, const int* in_sizes, int n_in,
                              void* d_out, int out_size, void* d_ws, size_t ws_size,
                              hipStream_t stream) {
  constexpr int Bc = 4, Sc = 1024, Dc = 1024, Hc = 16, Fc = 4096, Lc = 4;
  constexpr int M = Bc * Sc;  // 4096 rows

  const int*   tokens = (const int*)  d_in[0];
  const float* embedW = (const float*)d_in[1];
  const float* Wq = (const float*)d_in[2];
  const float* Wk = (const float*)d_in[3];
  const float* Wv = (const float*)d_in[4];
  const float* Wo = (const float*)d_in[5];
  const float* bq = (const float*)d_in[6];
  const float* bk = (const float*)d_in[7];
  const float* bv = (const float*)d_in[8];
  const float* bo = (const float*)d_in[9];
  const float* W1 = (const float*)d_in[10];
  const float* b1 = (const float*)d_in[11];
  const float* W2 = (const float*)d_in[12];
  const float* b2 = (const float*)d_in[13];
  const float* g1 = (const float*)d_in[14];
  const float* be1 = (const float*)d_in[15];
  const float* g2 = (const float*)d_in[16];
  const float* be2 = (const float*)d_in[17];
  const float* gf = (const float*)d_in[18];
  const float* bef = (const float*)d_in[19];

  // workspace carve-out (256B aligned)
  char* p = (char*)d_ws;
  auto alloc = [&](size_t bytes) -> void* {
    void* r = (void*)p;
    p += (bytes + 255) & ~(size_t)255;
    return r;
  };
  const size_t wDD = (size_t)Lc * Dc * Dc;
  const size_t wDF = (size_t)Lc * Dc * Fc;
  _Float16* wq16 = (_Float16*)alloc(wDD * 2);
  _Float16* wk16 = (_Float16*)alloc(wDD * 2);
  _Float16* wv16 = (_Float16*)alloc(wDD * 2);
  _Float16* wo16 = (_Float16*)alloc(wDD * 2);
  _Float16* w116 = (_Float16*)alloc(wDF * 2);
  _Float16* w216 = (_Float16*)alloc(wDF * 2);
  float*    x32  = (float*)   alloc((size_t)M * Dc * 4);
  _Float16* x16  = (_Float16*)alloc((size_t)M * Dc * 2);
  _Float16* q16  = (_Float16*)alloc((size_t)M * Dc * 2);
  _Float16* k16  = (_Float16*)alloc((size_t)M * Dc * 2);
  _Float16* v16  = (_Float16*)alloc((size_t)M * Dc * 2);
  _Float16* o16  = (_Float16*)alloc((size_t)M * Dc * 2);
  float*    tmp32 = (float*)  alloc((size_t)M * Dc * 4);
  _Float16* h16  = (_Float16*)alloc((size_t)M * Fc * 2);
  (void)ws_size; (void)n_in; (void)in_sizes; (void)out_size;

  // weights -> f16
  cvt_f32_f16<<<4096, 256, 0, stream>>>(Wq, wq16, wDD);
  cvt_f32_f16<<<4096, 256, 0, stream>>>(Wk, wk16, wDD);
  cvt_f32_f16<<<4096, 256, 0, stream>>>(Wv, wv16, wDD);
  cvt_f32_f16<<<4096, 256, 0, stream>>>(Wo, wo16, wDD);
  cvt_f32_f16<<<8192, 256, 0, stream>>>(W1, w116, wDF);
  cvt_f32_f16<<<8192, 256, 0, stream>>>(W2, w216, wDF);

  // embedding + positional encoding
  embed_pe<<<M, 256, 0, stream>>>(tokens, embedW, x32, x16, Sc, Dc);

  const dim3 gDD(Dc / 128, M / 128);   // 8 x 32
  const dim3 gDF(Fc / 128, M / 128);   // 32 x 32
  const dim3 gFD(Dc / 128, M / 128);   // 8 x 32 (K = F)
  const int attnBlocks = Bc * Hc * (Sc / 128);  // 512

  for (int l = 0; l < Lc; ++l) {
    const size_t oDD = (size_t)l * Dc * Dc;
    const size_t oDF = (size_t)l * Dc * Fc;
    // Q, K, V projections (f16 out for attention)
    gemm_wmma<0,0,1><<<gDD, 256, 0, stream>>>(x16, wq16 + oDD, bq + (size_t)l * Dc,
                                              nullptr, q16, M, Dc, Dc);
    gemm_wmma<0,0,1><<<gDD, 256, 0, stream>>>(x16, wk16 + oDD, bk + (size_t)l * Dc,
                                              nullptr, k16, M, Dc, Dc);
    gemm_wmma<0,0,1><<<gDD, 256, 0, stream>>>(x16, wv16 + oDD, bv + (size_t)l * Dc,
                                              nullptr, v16, M, Dc, Dc);
    // fused attention
    attn_flash<<<attnBlocks, 256, 0, stream>>>(q16, k16, v16, o16, Bc, Sc, Hc, Dc);
    // output projection -> f32, then residual + LN1
    gemm_wmma<0,1,0><<<gDD, 256, 0, stream>>>(o16, wo16 + oDD, bo + (size_t)l * Dc,
                                              tmp32, nullptr, M, Dc, Dc);
    residual_ln<<<M, 256, 0, stream>>>(x32, tmp32, g1 + (size_t)l * Dc,
                                       be1 + (size_t)l * Dc, x32, x16, Dc);
    // FFN: relu(x W1 + b1) W2 + b2, then residual + LN2
    gemm_wmma<1,0,1><<<gDF, 256, 0, stream>>>(x16, w116 + oDF, b1 + (size_t)l * Fc,
                                              nullptr, h16, M, Fc, Dc);
    gemm_wmma<0,1,0><<<gFD, 256, 0, stream>>>(h16, w216 + oDF, b2 + (size_t)l * Dc,
                                              tmp32, nullptr, M, Dc, Fc);
    residual_ln<<<M, 256, 0, stream>>>(x32, tmp32, g2 + (size_t)l * Dc,
                                       be2 + (size_t)l * Dc, x32, x16, Dc);
  }
  // final LayerNorm straight into d_out (f32)
  residual_ln<<<M, 256, 0, stream>>>(x32, nullptr, gf, bef, (float*)d_out, nullptr, Dc);
}
